// MeshGNN_39848706572312
// MI455X (gfx1250) — matrix operand order repaired and checked by reading, not compiled, not run on hardware
//
#include <hip/hip_runtime.h>
#include <hip/hip_bf16.h>

// ---------------------------------------------------------------------------
// MeshGNN (4-layer GCN) for MI455X / gfx1250, wave32.
//  - sparse propagate: edge-parallel gather/scale/atomic-scatter (fp32, L2-resident H)
//  - dense 256x256 layers: V_WMMA_F32_16X16X32_BF16, fp32 accumulate, bias+ReLU fused
//    64x256 block tile, A and W slabs staged in LDS as bf16, 8 WMMA per K-step/wave
// ---------------------------------------------------------------------------

typedef __attribute__((ext_vector_type(16))) __bf16 v16bf;
typedef __attribute__((ext_vector_type(8)))  float  v8f;

#define HID 256

// ---------------- graph preprocessing ----------------
__global__ void k_deg_init(float* deg, int n) {
  int i = blockIdx.x * blockDim.x + threadIdx.x;
  if (i < n) deg[i] = 1.0f;                         // self loop contributes 1
}

__global__ void k_deg_edges(const int* __restrict__ row, float* deg, int E) {
  int e = blockIdx.x * blockDim.x + threadIdx.x;
  if (e < E) atomicAdd(&deg[row[e]], 1.0f);
}

__global__ void k_dinv(const float* __restrict__ deg, float* dinv, int n) {
  int i = blockIdx.x * blockDim.x + threadIdx.x;
  if (i < n) { float d = deg[i]; dinv[i] = d > 0.0f ? rsqrtf(d) : 0.0f; }
}

__global__ void k_norm(const int* __restrict__ row, const int* __restrict__ col,
                       const float* __restrict__ dinv, float* norm, int E) {
  int e = blockIdx.x * blockDim.x + threadIdx.x;
  if (e < E) norm[e] = dinv[row[e]] * dinv[col[e]];
}

// ---------------- propagate, 3-wide features ----------------
__global__ void k_prop_self3(const float* __restrict__ dinv, const float* __restrict__ h,
                             float* out, int n) {
  int i = blockIdx.x * blockDim.x + threadIdx.x;
  if (i < n) {
    float d = dinv[i]; float w = d * d;
    out[i*3+0] = w * h[i*3+0];
    out[i*3+1] = w * h[i*3+1];
    out[i*3+2] = w * h[i*3+2];
  }
}

__global__ void k_prop_edges3(const int* __restrict__ row, const int* __restrict__ col,
                              const float* __restrict__ norm, const float* __restrict__ h,
                              float* out, int E) {
  int e = blockIdx.x * blockDim.x + threadIdx.x;
  if (e < E) {
    int r = row[e], c = col[e]; float w = norm[e];
    atomicAdd(&out[r*3+0], w * h[c*3+0]);
    atomicAdd(&out[r*3+1], w * h[c*3+1]);
    atomicAdd(&out[r*3+2], w * h[c*3+2]);
  }
}

// ---------------- propagate, 256-wide features ----------------
// self-loop pass WRITES the destination (initializes it), edge pass atomically adds.
__global__ void k_prop_self256(const float* __restrict__ dinv, const float* __restrict__ h,
                               float* out, int n) {
  int t = blockIdx.x * blockDim.x + threadIdx.x;
  int i = t >> 6;              // 64 threads per node
  int f = (t & 63) << 2;       // 4 features per thread
  if (i >= n) return;
  float d = dinv[i]; float w = d * d;
  const float4 v = *(const float4*)(h + (size_t)i * HID + f);
  float4 o = { w*v.x, w*v.y, w*v.z, w*v.w };
  *(float4*)(out + (size_t)i * HID + f) = o;
}

__global__ void k_prop_edges256(const int* __restrict__ row, const int* __restrict__ col,
                                const float* __restrict__ norm, const float* __restrict__ h,
                                float* out, int E) {
  int t = blockIdx.x * blockDim.x + threadIdx.x;
  int e = t >> 6;              // 64 threads per edge
  int f = (t & 63) << 2;       // 4 features per thread
  if (e >= E) return;
  int r = row[e], c = col[e]; float w = norm[e];
  const float4 v = *(const float4*)(h + (size_t)c * HID + f);
  float* dst = out + (size_t)r * HID + f;
  atomicAdd(dst + 0, w * v.x);
  atomicAdd(dst + 1, w * v.y);
  atomicAdd(dst + 2, w * v.z);
  atomicAdd(dst + 3, w * v.w);
}

// ---------------- layer 0: [N,3] @ [3,256] + b, relu (K=3: scalar) ----------------
__global__ void k_lin0(const float* __restrict__ p, const float* __restrict__ W0,
                       const float* __restrict__ b0, float* h, int n) {
  int t = blockIdx.x * blockDim.x + threadIdx.x;
  int i = t >> 8, j = t & 255;
  if (i >= n) return;
  float a = b0[j]
          + p[i*3+0] * W0[0*HID + j]
          + p[i*3+1] * W0[1*HID + j]
          + p[i*3+2] * W0[2*HID + j];
  h[(size_t)i * HID + j] = fmaxf(a, 0.0f);
}

// ---------------- hidden layers: [N,256] @ [256,256] + b, relu : bf16 WMMA ----------
// Block = 256 threads = 8 waves, computes a 64-row x 256-col output tile.
// Wave w owns column tiles [32w,32w+16) and [32w+16,32w+32) across ALL 4 row tiles
// (8 v8f accumulators). Per K-step of 32:
//   - W slab (32x256) staged in LDS as bf16, [col][k] order -> B fragment is 16
//     contiguous bf16 per lane (one 32B LDS read)
//   - A slab (64x32) staged in LDS as bf16, [row][k] order -> A fragment is two
//     contiguous 16B LDS reads per lane (K = half*16+i and half*16+8+i pattern)
//   - 8 WMMAs per wave per stage (B fragments reused across 4 row tiles)
__global__ void k_gemm256_wmma(const float* __restrict__ A, const float* __restrict__ W,
                               const float* __restrict__ bias, float* __restrict__ out,
                               int nrows, int do_relu) {
  __shared__ __bf16 ldsW[32 * HID];          // [col*32 + k'], 16 KB
  __shared__ __bf16 ldsA[64 * 32];           // [row*32 + k'],  4 KB

  const int rowbase = blockIdx.x * 64;
  const int wv   = threadIdx.x >> 5;
  const int lane = threadIdx.x & 31;
  const int half = lane >> 4;                // K-half selector (A/B layouts)
  const int m    = lane & 15;                // tile row (A) / tile column (B)
  const int n0a  = wv * 32;
  const int n0b  = wv * 32 + 16;

  // A staging assignment: thread t loads 8 consecutive K-values of one row
  const int arow_l = threadIdx.x >> 2;                  // 0..63 local row
  const int ak0    = (threadIdx.x & 3) * 8;             // 0,8,16,24
  const int arow_g = min(rowbase + arow_l, nrows - 1);  // clamp for tail safety
  const float* Arow = A + (size_t)arow_g * HID;

  v8f acc[4][2];
  #pragma unroll
  for (int rt = 0; rt < 4; ++rt) { acc[rt][0] = (v8f){}; acc[rt][1] = (v8f){}; }

  for (int kb = 0; kb < HID; kb += 32) {
    __syncthreads();                          // protect LDS reuse across K-steps
    {
      // stage W[kb..kb+32, 0..256) -> ldsW[col][k'], coalesced global reads
      int colw = threadIdx.x;                 // 0..255
      const float* wp = W + (size_t)kb * HID + colw;
      __bf16* lp = ldsW + colw * 32;
      #pragma unroll
      for (int kk = 0; kk < 32; ++kk)
        lp[kk] = (__bf16)wp[(size_t)kk * HID];

      // stage A[rowbase..rowbase+64, kb..kb+32) -> ldsA[row][k'] (8 floats/thread)
      const float* ap = Arow + kb + ak0;
      __bf16* la = ldsA + arow_l * 32 + ak0;
      #pragma unroll
      for (int kk = 0; kk < 8; ++kk)
        la[kk] = (__bf16)ap[kk];
    }
    __syncthreads();

    // B fragments: lane holds B[kb + half*16 + i, n0 + m], i = 0..15 (contiguous)
    v16bf b0v, b1v;
    {
      const __bf16* bp0 = ldsW + (n0a + m) * 32 + half * 16;
      const __bf16* bp1 = ldsW + (n0b + m) * 32 + half * 16;
      #pragma unroll
      for (int i = 0; i < 16; ++i) b0v[i] = bp0[i];
      #pragma unroll
      for (int i = 0; i < 16; ++i) b1v[i] = bp1[i];
    }

    #pragma unroll
    for (int rt = 0; rt < 4; ++rt) {
      // A fragment: lane holds K = half*16+i, i = 0..15 (contiguous, matches B)
      v16bf a;
      const __bf16* apl = ldsA + (rt * 16 + m) * 32 + half * 16;
      #pragma unroll
      for (int i = 0; i < 8; ++i) a[i] = apl[i];
      #pragma unroll
      for (int i = 0; i < 8; ++i) a[8 + i] = apl[8 + i];

      acc[rt][0] = __builtin_amdgcn_wmma_f32_16x16x32_bf16(false, a, false, b0v,
                                                           (short)0, acc[rt][0],
                                                           false, false);
      acc[rt][1] = __builtin_amdgcn_wmma_f32_16x16x32_bf16(false, a, false, b1v,
                                                           (short)0, acc[rt][1],
                                                           false, false);
    }
  }

  // C/D layout: VGPR r, lane L -> M = r + 8*(L/16), N = L%16
  const float bias0 = bias[n0a + m];
  const float bias1 = bias[n0b + m];
  #pragma unroll
  for (int rt = 0; rt < 4; ++rt) {
    #pragma unroll
    for (int r = 0; r < 8; ++r) {
      int orow = rowbase + rt * 16 + r + 8 * half;
      if (orow >= nrows) continue;
      float v0 = acc[rt][0][r] + bias0;
      float v1 = acc[rt][1][r] + bias1;
      if (do_relu) { v0 = fmaxf(v0, 0.0f); v1 = fmaxf(v1, 0.0f); }
      out[(size_t)orow * HID + n0a + m] = v0;
      out[(size_t)orow * HID + n0b + m] = v1;
    }
  }
}

// ---------------- final: p @ W3 + b3 + x @ Wr + br (wave per node) ----------------
__global__ void k_final(const float* __restrict__ p, const float* __restrict__ W3,
                        const float* __restrict__ b3, const float* __restrict__ x,
                        const float* __restrict__ Wr, const float* __restrict__ br,
                        float* out, int n) {
  int gt   = blockIdx.x * blockDim.x + threadIdx.x;
  int node = gt >> 5;
  int lane = threadIdx.x & 31;
  if (node >= n) return;

  const float* pr = p + (size_t)node * HID;
  float a0 = 0.f, a1 = 0.f, a2 = 0.f;
  for (int k = lane; k < HID; k += 32) {
    float v = pr[k];
    a0 += v * W3[k*3 + 0];
    a1 += v * W3[k*3 + 1];
    a2 += v * W3[k*3 + 2];
  }
  for (int off = 16; off > 0; off >>= 1) {   // wave32 reduction
    a0 += __shfl_down(a0, off, 32);
    a1 += __shfl_down(a1, off, 32);
    a2 += __shfl_down(a2, off, 32);
  }
  if (lane == 0) {
    float x0 = x[node*3+0], x1 = x[node*3+1], x2 = x[node*3+2];
    out[node*3+0] = a0 + b3[0] + x0*Wr[0] + x1*Wr[3] + x2*Wr[6] + br[0];
    out[node*3+1] = a1 + b3[1] + x0*Wr[1] + x1*Wr[4] + x2*Wr[7] + br[1];
    out[node*3+2] = a2 + b3[2] + x0*Wr[2] + x1*Wr[5] + x2*Wr[8] + br[2];
  }
}

// ---------------------------------------------------------------------------
extern "C" void kernel_launch(void* const* d_in, const int* in_sizes, int n_in,
                              void* d_out, int out_size, void* d_ws, size_t ws_size,
                              hipStream_t stream) {
  const float* x  = (const float*)d_in[0];
  const int*   ei = (const int*)  d_in[1];
  const float* W0 = (const float*)d_in[2];
  const float* b0 = (const float*)d_in[3];
  const float* W1 = (const float*)d_in[4];
  const float* b1 = (const float*)d_in[5];
  const float* W2 = (const float*)d_in[6];
  const float* b2 = (const float*)d_in[7];
  const float* W3 = (const float*)d_in[8];
  const float* b3 = (const float*)d_in[9];
  const float* Wr = (const float*)d_in[10];
  const float* br = (const float*)d_in[11];

  const int N = in_sizes[0] / 3;
  const int E = in_sizes[1] / 2;
  const int* row = ei;
  const int* col = ei + E;

  // workspace layout (floats): deg[N] dinv[N] norm[E] px[3N] hA[256N] hB[256N]
  float* ws   = (float*)d_ws;
  float* deg  = ws;
  float* dinv = deg  + N;
  float* norm = dinv + N;
  float* px   = norm + E;
  float* hA   = px + (size_t)N * 3;
  float* hB   = hA + (size_t)N * HID;
  float* out  = (float*)d_out;

  const int T = 256;
  auto cdiv = [](long long a, long long b) { return (int)((a + b - 1) / b); };

  // ---- normalization coefficients ----
  k_deg_init <<<cdiv(N, T), T, 0, stream>>>(deg, N);
  k_deg_edges<<<cdiv(E, T), T, 0, stream>>>(row, deg, E);
  k_dinv     <<<cdiv(N, T), T, 0, stream>>>(deg, dinv, N);
  k_norm     <<<cdiv(E, T), T, 0, stream>>>(row, col, dinv, norm, E);

  // ---- layer 0: h = relu(propagate(x) @ W0 + b0) ----
  k_prop_self3 <<<cdiv(N, T), T, 0, stream>>>(dinv, x, px, N);
  k_prop_edges3<<<cdiv(E, T), T, 0, stream>>>(row, col, norm, x, px, E);
  k_lin0<<<cdiv((long long)N * HID, T), T, 0, stream>>>(px, W0, b0, hA, N);

  // ---- layer 1: h = relu(propagate(h) @ W1 + b1) ----
  k_prop_self256 <<<cdiv((long long)N * 64, T), T, 0, stream>>>(dinv, hA, hB, N);
  k_prop_edges256<<<cdiv((long long)E * 64, T), T, 0, stream>>>(row, col, norm, hA, hB, E);
  k_gemm256_wmma <<<cdiv(N, 64), T, 0, stream>>>(hB, W1, b1, hA, N, 1);

  // ---- layer 2: h = relu(propagate(h) @ W2 + b2) ----
  k_prop_self256 <<<cdiv((long long)N * 64, T), T, 0, stream>>>(dinv, hA, hB, N);
  k_prop_edges256<<<cdiv((long long)E * 64, T), T, 0, stream>>>(row, col, norm, hA, hB, E);
  k_gemm256_wmma <<<cdiv(N, 64), T, 0, stream>>>(hB, W2, b2, hA, N, 1);

  // ---- layer 3 + residual: out = propagate(h) @ W3 + b3 + x @ Wr + br ----
  k_prop_self256 <<<cdiv((long long)N * 64, T), T, 0, stream>>>(dinv, hA, hB, N);
  k_prop_edges256<<<cdiv((long long)E * 64, T), T, 0, stream>>>(row, col, norm, hA, hB, E);
  k_final<<<cdiv((long long)N * 32, T), T, 0, stream>>>(hB, W3, b3, x, Wr, br, out, N);
}
